// CopyLSTMDecoder_24730421690598
// MI455X (gfx1250) — matrix operand order repaired and checked by reading, not compiled, not run on hardware
//
#include <hip/hip_runtime.h>
#include <hip/hip_fp16.h>
#include <stdint.h>
#include <stddef.h>

#define BB 32
#define TT 48
#define LL 400
#define VV 32000
#define EE 256
#define HH 256
#define EXTV 32050

typedef __attribute__((ext_vector_type(16))) _Float16 v16h;
typedef __attribute__((ext_vector_type(8)))  float    v8f;

__device__ __forceinline__ float sigf(float x) { return 1.0f / (1.0f + __expf(-x)); }

// ---------------- one-time: embedding f32 -> f16 ----------------
__global__ void k_convert(const float* __restrict__ emb, _Float16* __restrict__ embH) {
    int i = blockIdx.x * 256 + threadIdx.x;           // V*E total
    embH[i] = (_Float16)emb[i];
}

// ---------------- LSTM cell 0: x = [emb(tok), prev_out] (512), h0(256) ----------------
__global__ void k_cell0(const int* __restrict__ abstract, const float* __restrict__ embedding,
                        const float* __restrict__ pout,
                        const float* __restrict__ hcur, const float* __restrict__ ccur,
                        const float* __restrict__ Wih, const float* __restrict__ Whh,
                        const float* __restrict__ bih, const float* __restrict__ bhh,
                        float* __restrict__ hnxt, float* __restrict__ cnxt, int t) {
    int b = blockIdx.x, j = threadIdx.x;
    int tok = abstract[b * TT + t];
    float si = 0.f, sf = 0.f, sg = 0.f, so = 0.f;
    const float* er = embedding + (size_t)tok * EE;
    const float* pr = pout + (size_t)b * EE;
    const float* wi0 = Wih + (size_t)(0 * HH + j) * 512;
    const float* wi1 = Wih + (size_t)(1 * HH + j) * 512;
    const float* wi2 = Wih + (size_t)(2 * HH + j) * 512;
    const float* wi3 = Wih + (size_t)(3 * HH + j) * 512;
    #pragma unroll 4
    for (int k = 0; k < EE; ++k) {
        float xv = er[k];
        si = fmaf(xv, wi0[k], si); sf = fmaf(xv, wi1[k], sf);
        sg = fmaf(xv, wi2[k], sg); so = fmaf(xv, wi3[k], so);
    }
    #pragma unroll 4
    for (int k = 0; k < EE; ++k) {
        float xv = pr[k];
        si = fmaf(xv, wi0[EE + k], si); sf = fmaf(xv, wi1[EE + k], sf);
        sg = fmaf(xv, wi2[EE + k], sg); so = fmaf(xv, wi3[EE + k], so);
    }
    const float* hr = hcur + (size_t)b * HH;          // layer 0
    const float* wh0 = Whh + (size_t)(0 * HH + j) * HH;
    const float* wh1 = Whh + (size_t)(1 * HH + j) * HH;
    const float* wh2 = Whh + (size_t)(2 * HH + j) * HH;
    const float* wh3 = Whh + (size_t)(3 * HH + j) * HH;
    #pragma unroll 4
    for (int k = 0; k < HH; ++k) {
        float hv = hr[k];
        si = fmaf(hv, wh0[k], si); sf = fmaf(hv, wh1[k], sf);
        sg = fmaf(hv, wh2[k], sg); so = fmaf(hv, wh3[k], so);
    }
    float gi = si + bih[0 * HH + j] + bhh[0 * HH + j];
    float gf = sf + bih[1 * HH + j] + bhh[1 * HH + j];
    float gg = sg + bih[2 * HH + j] + bhh[2 * HH + j];
    float go = so + bih[3 * HH + j] + bhh[3 * HH + j];
    float cold = ccur[(size_t)b * HH + j];            // layer 0
    float cn = sigf(gf) * cold + sigf(gi) * tanhf(gg);
    float hn = sigf(go) * tanhf(cn);
    hnxt[(size_t)b * HH + j] = hn;
    cnxt[(size_t)b * HH + j] = cn;
}

// ---------------- LSTM cell 1: h0n(256), h1(256) ----------------
__global__ void k_cell1(const float* __restrict__ h0n, const float* __restrict__ hcur,
                        const float* __restrict__ ccur,
                        const float* __restrict__ Wih, const float* __restrict__ Whh,
                        const float* __restrict__ bih, const float* __restrict__ bhh,
                        float* __restrict__ hnxt, float* __restrict__ cnxt) {
    int b = blockIdx.x, j = threadIdx.x;
    float si = 0.f, sf = 0.f, sg = 0.f, so = 0.f;
    const float* xr = h0n + (size_t)b * HH;           // new layer-0 hidden
    const float* hr = hcur + (size_t)BB * HH + (size_t)b * HH;   // old layer-1 hidden
    const float* wi0 = Wih + (size_t)(0 * HH + j) * HH;
    const float* wi1 = Wih + (size_t)(1 * HH + j) * HH;
    const float* wi2 = Wih + (size_t)(2 * HH + j) * HH;
    const float* wi3 = Wih + (size_t)(3 * HH + j) * HH;
    const float* wh0 = Whh + (size_t)(0 * HH + j) * HH;
    const float* wh1 = Whh + (size_t)(1 * HH + j) * HH;
    const float* wh2 = Whh + (size_t)(2 * HH + j) * HH;
    const float* wh3 = Whh + (size_t)(3 * HH + j) * HH;
    #pragma unroll 4
    for (int k = 0; k < HH; ++k) {
        float xv = xr[k], hv = hr[k];
        si = fmaf(xv, wi0[k], si); si = fmaf(hv, wh0[k], si);
        sf = fmaf(xv, wi1[k], sf); sf = fmaf(hv, wh1[k], sf);
        sg = fmaf(xv, wi2[k], sg); sg = fmaf(hv, wh2[k], sg);
        so = fmaf(xv, wi3[k], so); so = fmaf(hv, wh3[k], so);
    }
    float gi = si + bih[0 * HH + j] + bhh[0 * HH + j];
    float gf = sf + bih[1 * HH + j] + bhh[1 * HH + j];
    float gg = sg + bih[2 * HH + j] + bhh[2 * HH + j];
    float go = so + bih[3 * HH + j] + bhh[3 * HH + j];
    float cold = ccur[(size_t)BB * HH + (size_t)b * HH + j];
    float cn = sigf(gf) * cold + sigf(gi) * tanhf(gg);
    float hn = sigf(go) * tanhf(cn);
    hnxt[(size_t)BB * HH + (size_t)b * HH + j] = hn;
    cnxt[(size_t)BB * HH + (size_t)b * HH + j] = cn;
}

// ---------------- query = h1 @ attn_w ----------------
__global__ void k_query(const float* __restrict__ hnxt, const float* __restrict__ attn_w,
                        float* __restrict__ query) {
    int b = blockIdx.x, d = threadIdx.x;
    const float* hr = hnxt + (size_t)BB * HH + (size_t)b * HH;
    float s = 0.f;
    #pragma unroll 4
    for (int k = 0; k < HH; ++k) s = fmaf(hr[k], attn_w[(size_t)k * HH + d], s);
    query[(size_t)b * HH + d] = s;
}

// ---------------- scores[b,l] = query[b]·enc_proj[b,l] ----------------
__global__ void k_score(const float* __restrict__ query, const float* __restrict__ enc_proj,
                        float* __restrict__ scores) {
    int idx = blockIdx.x * 256 + threadIdx.x;
    if (idx >= BB * LL) return;
    int b = idx / LL, l = idx % LL;
    const float* q = query + (size_t)b * HH;
    const float* p = enc_proj + ((size_t)b * LL + l) * HH;
    float s = 0.f;
    #pragma unroll 4
    for (int k = 0; k < HH; ++k) s = fmaf(q[k], p[k], s);
    scores[idx] = s;
}

// ---------------- mask + softmax over L + ctx + copy gate ----------------
__global__ void k_softmax_ctx(const float* __restrict__ scores, const unsigned char* __restrict__ mask,
                              const float* __restrict__ enc_mem, float* __restrict__ attn,
                              float* __restrict__ ctx,
                              const float* __restrict__ hnxt,
                              const int* __restrict__ abstract, const float* __restrict__ embedding,
                              const float* __restrict__ v_c, const float* __restrict__ v_s,
                              const float* __restrict__ v_i, const float* __restrict__ copy_b,
                              float* __restrict__ gate, int t) {
    int b = blockIdx.x, tid = threadIdx.x;
    __shared__ float sc[LL];
    __shared__ float red[256];
    __shared__ int anyvalid;
    int localAny = 0;
    for (int l = tid; l < LL; l += 256) {
        int m = mask[b * LL + l] ? 1 : 0;
        localAny |= m;
        sc[l] = m ? scores[b * LL + l] : -1e18f;
    }
    red[tid] = (float)localAny;
    __syncthreads();
    for (int s = 128; s > 0; s >>= 1) { if (tid < s) red[tid] = fmaxf(red[tid], red[tid + s]); __syncthreads(); }
    if (tid == 0) anyvalid = (red[0] > 0.f);
    __syncthreads();
    if (!anyvalid) { for (int l = tid; l < LL; l += 256) sc[l] = 0.f; __syncthreads(); }
    // max
    float mx = -1e30f;
    for (int l = tid; l < LL; l += 256) mx = fmaxf(mx, sc[l]);
    red[tid] = mx; __syncthreads();
    for (int s = 128; s > 0; s >>= 1) { if (tid < s) red[tid] = fmaxf(red[tid], red[tid + s]); __syncthreads(); }
    mx = red[0]; __syncthreads();
    // sum
    float sm = 0.f;
    for (int l = tid; l < LL; l += 256) sm += __expf(sc[l] - mx);
    red[tid] = sm; __syncthreads();
    for (int s = 128; s > 0; s >>= 1) { if (tid < s) red[tid] += red[tid + s]; __syncthreads(); }
    sm = red[0]; __syncthreads();
    float inv = 1.0f / sm;
    for (int l = tid; l < LL; l += 256) {
        float a = __expf(sc[l] - mx) * inv;
        sc[l] = a;
        attn[b * LL + l] = a;
    }
    __syncthreads();
    // ctx[b, d]
    int d = tid;
    float c = 0.f;
    const float* em = enc_mem + (size_t)b * LL * HH + d;
    for (int l = 0; l < LL; ++l) c = fmaf(sc[l], em[(size_t)l * HH], c);
    ctx[(size_t)b * HH + d] = c;
    // copy gate: ctx·v_c + h1·v_s + emb·v_i + copy_b
    int tok = abstract[b * TT + t];
    float h1d = hnxt[(size_t)BB * HH + (size_t)b * HH + d];
    float p = c * v_c[d] + h1d * v_s[d] + embedding[(size_t)tok * EE + d] * v_i[d];
    red[tid] = p; __syncthreads();
    for (int s = 128; s > 0; s >>= 1) { if (tid < s) red[tid] += red[tid + s]; __syncthreads(); }
    if (tid == 0) gate[b] = sigf(red[0] + copy_b[0]);
}

// ---------------- dec_out = [h1, ctx] @ proj_W.T + proj_b  (f32 + f16 copies) ----------------
__global__ void k_decout(const float* __restrict__ hnxt, const float* __restrict__ ctx,
                         const float* __restrict__ projW, const float* __restrict__ projb,
                         float* __restrict__ dec, _Float16* __restrict__ decH) {
    int b = blockIdx.x, e = threadIdx.x;
    const float* hr = hnxt + (size_t)BB * HH + (size_t)b * HH;
    const float* cr = ctx + (size_t)b * HH;
    const float* w = projW + (size_t)e * 512;
    float s = projb[e];
    #pragma unroll 4
    for (int k = 0; k < HH; ++k) s = fmaf(hr[k], w[k], s);
    #pragma unroll 4
    for (int k = 0; k < HH; ++k) s = fmaf(cr[k], w[HH + k], s);
    dec[(size_t)b * EE + e] = s;
    decH[(size_t)b * EE + e] = (_Float16)s;
}

// ---------------- logits = dec_out @ embedding.T  (WMMA f16) ----------------
// 250 blocks x 256 threads; each wave owns one 16-col tile and BOTH 16-row tiles
// (2 accumulators) so every B fragment is loaded exactly once.
__global__ void k_logits_wmma(const _Float16* __restrict__ decH, const _Float16* __restrict__ embH,
                              float* __restrict__ logits) {
    int wave = threadIdx.x >> 5;
    int lane = threadIdx.x & 31;
    int ntile = blockIdx.x * 8 + wave;                // 0..1999
    int n0 = ntile * 16;
    int rlo = lane & 15;
    int col = n0 + rlo;
    int abase = (lane < 16) ? 0 : 16;   // A: rows lane-striped, K halves split 0-15 | 16-31
    int bbase = (lane < 16) ? 0 : 16;   // B: columns lane-striped, K halves split 0-15 | 16-31
    v8f acc0 = {}, acc1 = {};
    for (int k0 = 0; k0 < EE; k0 += 32) {
        const _Float16* bp = embH + (size_t)col * EE + k0 + bbase;   // 32B aligned
        v16h bfr = *(const v16h*)bp;
        v16h a0, a1;
        const _Float16* ap0 = decH + (size_t)rlo * EE + k0 + abase;
        const _Float16* ap1 = decH + (size_t)(16 + rlo) * EE + k0 + abase;
        #pragma unroll
        for (int j = 0; j < 8; ++j) { a0[j] = ap0[j]; a0[8 + j] = ap0[8 + j]; }
        #pragma unroll
        for (int j = 0; j < 8; ++j) { a1[j] = ap1[j]; a1[8 + j] = ap1[8 + j]; }
        acc0 = __builtin_amdgcn_wmma_f32_16x16x32_f16(false, a0, false, bfr,
                                                      (short)0, acc0, false, false);
        acc1 = __builtin_amdgcn_wmma_f32_16x16x32_f16(false, a1, false, bfr,
                                                      (short)0, acc1, false, false);
    }
    int mbase = (lane < 16) ? 0 : 8;
    #pragma unroll
    for (int i = 0; i < 8; ++i) {
        logits[(size_t)(mbase + i) * VV + col]        = acc0[i];
        logits[(size_t)(16 + mbase + i) * VV + col]   = acc1[i];
    }
}

// ---------------- per-row max & sum(exp) over V ----------------
__global__ void k_rowstats(const float* __restrict__ logits, float* __restrict__ mx,
                           float* __restrict__ sm) {
    int b = blockIdx.x, tid = threadIdx.x;
    __shared__ float red[256];
    const float* row = logits + (size_t)b * VV;
    float m = -1e30f;
    for (int v = tid; v < VV; v += 256) m = fmaxf(m, row[v]);
    red[tid] = m; __syncthreads();
    for (int s = 128; s > 0; s >>= 1) { if (tid < s) red[tid] = fmaxf(red[tid], red[tid + s]); __syncthreads(); }
    m = red[0]; __syncthreads();
    float s = 0.f;
    for (int v = tid; v < VV; v += 256) s += __expf(row[v] - m);
    red[tid] = s; __syncthreads();
    for (int st = 128; st > 0; st >>= 1) { if (tid < st) red[tid] += red[tid + st]; __syncthreads(); }
    if (tid == 0) { mx[b] = m; sm[b] = red[0]; }
}

// ---------------- (1-g)*gen_prob into output slice ----------------
__global__ void k_final(const float* __restrict__ logits, const float* __restrict__ mx,
                        const float* __restrict__ sm, const float* __restrict__ gate,
                        float* __restrict__ out, int t) {
    int idx = blockIdx.x * 256 + threadIdx.x;
    if (idx >= BB * EXTV) return;
    int b = idx / EXTV, v = idx % EXTV;
    float gp = 0.f;
    if (v < VV) gp = __expf(logits[(size_t)b * VV + v] - mx[b]) / sm[b];
    out[(size_t)b * TT * EXTV + (size_t)t * EXTV + v] = (1.0f - gate[b]) * gp;
}

// ---------------- scatter copy probabilities ----------------
__global__ void k_scatter(const float* __restrict__ attn, const float* __restrict__ gate,
                          const int* __restrict__ extend_art, float* __restrict__ out, int t) {
    int idx = blockIdx.x * 256 + threadIdx.x;
    if (idx >= BB * LL) return;
    int b = idx / LL, l = idx % LL;
    int v = extend_art[b * LL + l];
    atomicAdd(out + (size_t)b * TT * EXTV + (size_t)t * EXTV + v, attn[idx] * gate[b]);
}

// ---------------- log ----------------
__global__ void k_log(float* __restrict__ out, int t) {
    int idx = blockIdx.x * 256 + threadIdx.x;
    if (idx >= BB * EXTV) return;
    int b = idx / EXTV, v = idx % EXTV;
    size_t off = (size_t)b * TT * EXTV + (size_t)t * EXTV + v;
    out[off] = logf(out[off] + 1e-12f);
}

extern "C" void kernel_launch(void* const* d_in, const int* in_sizes, int n_in,
                              void* d_out, int out_size, void* d_ws, size_t ws_size,
                              hipStream_t stream) {
    const int*   abstract   = (const int*)d_in[0];
    const float* enc_mem    = (const float*)d_in[1];
    const float* enc_proj   = (const float*)d_in[2];
    const unsigned char* mask = (const unsigned char*)d_in[3];
    const int*   extend_art = (const int*)d_in[4];
    const float* h0         = (const float*)d_in[6];
    const float* c0         = (const float*)d_in[7];
    const float* prev_out   = (const float*)d_in[8];
    const float* embedding  = (const float*)d_in[9];
    const float* W_ih0      = (const float*)d_in[10];
    const float* W_hh0      = (const float*)d_in[11];
    const float* b_ih0      = (const float*)d_in[12];
    const float* b_hh0      = (const float*)d_in[13];
    const float* W_ih1      = (const float*)d_in[14];
    const float* W_hh1      = (const float*)d_in[15];
    const float* b_ih1      = (const float*)d_in[16];
    const float* b_hh1      = (const float*)d_in[17];
    const float* attn_w     = (const float*)d_in[18];
    const float* proj_W     = (const float*)d_in[19];
    const float* proj_b     = (const float*)d_in[20];
    const float* v_c        = (const float*)d_in[21];
    const float* v_s        = (const float*)d_in[22];
    const float* v_i        = (const float*)d_in[23];
    const float* copy_b     = (const float*)d_in[24];
    float* out = (float*)d_out;

    // workspace carve-up (256B aligned blocks)
    char* base = (char*)d_ws;
    size_t off = 0;
    auto walloc = [&](size_t bytes) -> void* {
        void* p = base + off;
        off += (bytes + 255) & ~(size_t)255;
        return p;
    };
    _Float16* embH = (_Float16*)walloc((size_t)VV * EE * sizeof(_Float16));
    float* logits  = (float*)walloc((size_t)BB * VV * sizeof(float));
    float* hbuf[2] = { (float*)walloc(2 * BB * HH * sizeof(float)),
                       (float*)walloc(2 * BB * HH * sizeof(float)) };
    float* cbuf[2] = { (float*)walloc(2 * BB * HH * sizeof(float)),
                       (float*)walloc(2 * BB * HH * sizeof(float)) };
    float* dbuf[2] = { (float*)walloc((size_t)BB * EE * sizeof(float)),
                       (float*)walloc((size_t)BB * EE * sizeof(float)) };
    _Float16* decH = (_Float16*)walloc((size_t)BB * EE * sizeof(_Float16));
    float* query  = (float*)walloc((size_t)BB * HH * sizeof(float));
    float* scores = (float*)walloc((size_t)BB * LL * sizeof(float));
    float* attn   = (float*)walloc((size_t)BB * LL * sizeof(float));
    float* ctx    = (float*)walloc((size_t)BB * HH * sizeof(float));
    float* gate   = (float*)walloc(BB * sizeof(float));
    float* rmx    = (float*)walloc(BB * sizeof(float));
    float* rsm    = (float*)walloc(BB * sizeof(float));

    // init recurrent state
    hipMemcpyAsync(hbuf[0], h0, 2 * BB * HH * sizeof(float), hipMemcpyDeviceToDevice, stream);
    hipMemcpyAsync(cbuf[0], c0, 2 * BB * HH * sizeof(float), hipMemcpyDeviceToDevice, stream);
    hipMemcpyAsync(dbuf[0], prev_out, (size_t)BB * EE * sizeof(float), hipMemcpyDeviceToDevice, stream);

    k_convert<<<(VV * EE) / 256, 256, 0, stream>>>(embedding, embH);

    for (int t = 0; t < TT; ++t) {
        int cur = t & 1, nxt = cur ^ 1;
        k_cell0<<<BB, HH, 0, stream>>>(abstract, embedding, dbuf[cur], hbuf[cur], cbuf[cur],
                                       W_ih0, W_hh0, b_ih0, b_hh0, hbuf[nxt], cbuf[nxt], t);
        k_cell1<<<BB, HH, 0, stream>>>(hbuf[nxt], hbuf[cur], cbuf[cur], W_ih1, W_hh1, b_ih1, b_hh1,
                                       hbuf[nxt], cbuf[nxt]);
        k_query<<<BB, HH, 0, stream>>>(hbuf[nxt], attn_w, query);
        k_score<<<(BB * LL + 255) / 256, 256, 0, stream>>>(query, enc_proj, scores);
        k_softmax_ctx<<<BB, 256, 0, stream>>>(scores, mask, enc_mem, attn, ctx, hbuf[nxt],
                                              abstract, embedding, v_c, v_s, v_i, copy_b, gate, t);
        k_decout<<<BB, EE, 0, stream>>>(hbuf[nxt], ctx, proj_W, proj_b, dbuf[nxt], decH);
        k_logits_wmma<<<VV / 128, 256, 0, stream>>>(decH, embH, logits);
        k_rowstats<<<BB, 256, 0, stream>>>(logits, rmx, rsm);
        k_final<<<(BB * EXTV + 255) / 256, 256, 0, stream>>>(logits, rmx, rsm, gate, out, t);
        k_scatter<<<(BB * LL + 255) / 256, 256, 0, stream>>>(attn, gate, extend_art, out, t);
        k_log<<<(BB * EXTV + 255) / 256, 256, 0, stream>>>(out, t);
    }
}
